// SigMMD_15195594293657
// MI455X (gfx1250) — matrix operand order repaired and checked by reading, not compiled, not run on hardware
//
#include <hip/hip_runtime.h>
#include <hip/hip_bf16.h>

// ---------------------------------------------------------------------------
// Signature-kernel MMD on gfx1250 (MI455X).
//   X, Y : [64][128][64] f32.  Output: scalar f32.
//   Pipeline:
//     1) diff kernels: dXh/dYh = f16 path increments, [64][128][64] halves
//        (row 127 zero-padded -> extra Goursat row is the identity).
//     2) sig kernel: 12288 blocks (3 Grams x 64 x 64 pairs), 128 threads.
//        Per block: stage dX[a], dY[b] in LDS, compute the 128x128 increment
//        inner-product tile in 16-row panels with v_wmma_f32_16x16x32_f16,
//        then run the Goursat recurrence as 128 rows of (elementwise + block
//        prefix scan). Writes K[P][Q] per pair to workspace.
//     3) final reduce: deterministic weighted sum + initial-point MSE.
// ---------------------------------------------------------------------------

typedef _Float16 v16h __attribute__((ext_vector_type(16)));
typedef float    v8f  __attribute__((ext_vector_type(8)));

#define NA      64      // paths per bundle
#define TLEN    128     // time steps
#define DD      64      // feature dim (= WMMA K after 2 chained 16x16x32)
#define PQ      127     // increment count per path
#define COLS    128     // padded column count
#define ROWSTR  72      // padded LDS row stride in halves (bank-friendly, 16B aligned)
#define NPAIRS  (3 * NA * NA)   // 12288

union AFrag { v16h v; float f[8]; };

// ---- 1) path increments -> f16, padded -------------------------------------
__global__ void SigMMD_diff_kernel(const float* __restrict__ X,
                                   _Float16* __restrict__ dXh) {
    int idx = blockIdx.x * blockDim.x + threadIdx.x;   // over 64*128*64
    if (idx >= NA * TLEN * DD) return;
    int d = idx & (DD - 1);
    int p = (idx >> 6) & (TLEN - 1);
    int a = idx >> 13;
    float v = 0.0f;
    if (p < PQ) {
        const float* row = X + ((size_t)a * TLEN + p) * DD + d;
        v = row[DD] - row[0];
    }
    dXh[idx] = (_Float16)v;
}

// ---- 2) fused WMMA Gram tile + Goursat PDE per pair ------------------------
__global__ __launch_bounds__(128)
void SigMMD_sig_kernel(const _Float16* __restrict__ dXh,
                       const _Float16* __restrict__ dYh,
                       float* __restrict__ Kres) {
    __shared__ __align__(16) _Float16 Ash[TLEN * ROWSTR];  // dX[a], padded rows
    __shared__ __align__(16) _Float16 Bsh[TLEN * ROWSTR];  // dY[b], padded rows
    __shared__ __align__(16) float    Sinc[16][132];       // one 16x128 inc panel
    __shared__ float kbuf[2][COLS];                        // Goursat row ping-pong
    __shared__ float wsum[4];                              // per-wave scan totals

    const int tid  = threadIdx.x;       // 0..127
    const int lane = tid & 31;
    const int wid  = tid >> 5;          // 0..3
    const int m    = lane & 15;
    const int hi   = lane >> 4;         // 0: lanes 0-15, 1: lanes 16-31

    // pair decode: gram g (0:XX 1:YY 2:XY), row a, col b
    const int gid = blockIdx.x;
    const int g   = gid >> 12;          // /4096
    const int r   = gid & 4095;
    const int a   = r >> 6;
    const int b   = r & 63;
    const _Float16* __restrict__ srcA = (g == 1) ? dYh : dXh;
    const _Float16* __restrict__ srcB = (g == 0) ? dXh : dYh;

    // stage A and B operands into LDS (one padded row per thread)
    {
        const uint4* ga = reinterpret_cast<const uint4*>(srcA + (size_t)a * TLEN * DD)
                        + tid * (DD * 2 / 16);                 // 8 uint4 per row
        const uint4* gb = reinterpret_cast<const uint4*>(srcB + (size_t)b * TLEN * DD)
                        + tid * (DD * 2 / 16);
        uint4* la = reinterpret_cast<uint4*>(&Ash[tid * ROWSTR]);
        uint4* lb = reinterpret_cast<uint4*>(&Bsh[tid * ROWSTR]);
#pragma unroll
        for (int j = 0; j < 8; ++j) { la[j] = ga[j]; lb[j] = gb[j]; }
    }
    kbuf[0][tid] = 1.0f;                 // Goursat boundary K[0][:] = 1
    int pp = 0;
    __syncthreads();

    for (int t = 0; t < 8; ++t) {        // 8 panels of 16 PDE rows
        const int p0 = t * 16;

        // --- WMMA phase: Sinc[0..15][0..127] = dX rows p0..p0+15 x dY^T ---
        // A fragments (16x32 f16 layout, ISA 7.12.2), shared by both tiles.
        AFrag afr[2];
#pragma unroll
        for (int s = 0; s < 2; ++s) {
            const _Float16* arow = &Ash[(p0 + m) * ROWSTR];
#pragma unroll
            for (int j = 0; j < 8; ++j) {
                int k = 2 * j + ((j >= 4) ? 8 : 0) + 8 * hi + 32 * s;
                afr[s].f[j] = *reinterpret_cast<const float*>(&arow[k]);
            }
        }
#pragma unroll
        for (int tt = 0; tt < 2; ++tt) { // each wave owns 2 of 8 column tiles
            const int ct  = wid * 2 + tt;
            const int col = ct * 16 + m;
            const _Float16* brow = &Bsh[col * ROWSTR];
            v8f acc = {};
#pragma unroll
            for (int s = 0; s < 2; ++s) {
                AFrag bfr;
#pragma unroll
                for (int j = 0; j < 8; ++j) {
                    int d = 2 * j + ((j >= 4) ? 8 : 0) + 8 * hi + 32 * s;
                    bfr.f[j] = *reinterpret_cast<const float*>(&brow[d]);
                }
                acc = __builtin_amdgcn_wmma_f32_16x16x32_f16(
                        false, afr[s].v, false, bfr.v,
                        (short)0, acc, false, false);
            }
            // C/D layout: VGPR r -> row r + 8*hi, column = lane&15 (+tile)
#pragma unroll
            for (int rr = 0; rr < 8; ++rr)
                Sinc[rr + 8 * hi][col] = acc[rr];
        }
        __syncthreads();

        // --- Goursat phase: 16 rows, each = elementwise + block prefix scan
        for (int rl = 0; rl < 16; ++rl) {
            float kp = kbuf[pp][tid];
            float c  = 0.0f;
            if (tid > 0) {
                float km = kbuf[pp][tid - 1];
                c = kp - km + Sinc[rl][tid - 1] * km;
            }
            // inclusive scan: wave32 shuffle scan + cross-wave offsets
            float v = c;
#pragma unroll
            for (int dlt = 1; dlt < 32; dlt <<= 1) {
                float tv = __shfl_up(v, dlt, 32);
                if (lane >= dlt) v += tv;
            }
            if (lane == 31) wsum[wid] = v;
            __syncthreads();
            float off = 0.0f;
            for (int w2 = 0; w2 < wid; ++w2) off += wsum[w2];
            kbuf[pp ^ 1][tid] = 1.0f + v + off;
            __syncthreads();
            pp ^= 1;
        }
    }
    if (tid == 0) Kres[gid] = kbuf[pp][COLS - 1];   // K[P][Q]
}

// ---- 3) deterministic reduction -------------------------------------------
__global__ void SigMMD_reduce_kernel(const float* __restrict__ X,
                                     const float* __restrict__ Y,
                                     const float* __restrict__ Kres,
                                     float* __restrict__ out) {
    __shared__ float red[256];
    const int tid = threadIdx.x;
    float s = 0.0f;
    for (int i = tid; i < NPAIRS; i += 256) {
        float w = (i >= 2 * NA * NA) ? (-2.0f / 4096.0f) : (1.0f / 4096.0f);
        s += w * Kres[i];
    }
    for (int i = tid; i < NA * DD; i += 256) {       // mean((X0 - Y0)^2)
        int a = i >> 6, d = i & 63;
        float df = X[(size_t)a * TLEN * DD + d] - Y[(size_t)a * TLEN * DD + d];
        s += df * df * (1.0f / 4096.0f);
    }
    red[tid] = s;
    __syncthreads();
    for (int st = 128; st > 0; st >>= 1) {
        if (tid < st) red[tid] += red[tid + st];
        __syncthreads();
    }
    if (tid == 0) out[0] = red[0];
}

// ---------------------------------------------------------------------------
extern "C" void kernel_launch(void* const* d_in, const int* in_sizes, int n_in,
                              void* d_out, int out_size, void* d_ws, size_t ws_size,
                              hipStream_t stream) {
    const float* X = (const float*)d_in[0];
    const float* Y = (const float*)d_in[1];
    float* out = (float*)d_out;

    _Float16* dXh = (_Float16*)d_ws;
    _Float16* dYh = dXh + (size_t)NA * TLEN * DD;
    float*    Kres = (float*)(dYh + (size_t)NA * TLEN * DD);

    const int nElem = NA * TLEN * DD;
    SigMMD_diff_kernel<<<(nElem + 255) / 256, 256, 0, stream>>>(X, dXh);
    SigMMD_diff_kernel<<<(nElem + 255) / 256, 256, 0, stream>>>(Y, dYh);
    SigMMD_sig_kernel<<<NPAIRS, 128, 0, stream>>>(dXh, dYh, Kres);
    SigMMD_reduce_kernel<<<1, 256, 0, stream>>>(X, Y, Kres, out);
}
